// WeatherDifferentiatedAttention_45853070852148
// MI455X (gfx1250) — matrix operand order repaired and checked by reading, not compiled, not run on hardware
//
#include <hip/hip_runtime.h>

#define D_MODEL 256
#define D_HID   64
#define NT      200
#define NW      100
#define NTILES_T 13
#define NTILES_W 7
#define LN_EPS  1e-5f

typedef __attribute__((ext_vector_type(16))) __bf16 v16bf;
typedef __attribute__((ext_vector_type(8)))  float  v8f;

struct alignas(16) U128 { unsigned int x[4]; };
union FragAB { v16bf v; __bf16 h[16]; U128 q[2]; };
union Pack8  { __bf16 h[8]; U128 q; };

// ---- LDS layout (bytes) -----------------------------------------------------
// OFF_K region is time-shared: h_wT [64][112] (phases W/A) -> k [208][64] (B/C)
#define OFF_W0   0         // 32768: w_w -> w_t -> (in_proj|attn_out) -> out_proj
#define OFF_HW   32768     // 14336: h_w [112][64] bf16 (row major)
#define OFF_Q    47104     // 26624: ctx -> q -> enh [208][64] bf16
#define OFF_K    73728     // 26624: h_wT [64][112] then k [208][64] bf16
#define OFF_V    100352    // 26624: vT [64][208] bf16 (transposed!)
#define OFF_STG  126976    // 8 waves * 3072 (2KB [16][64] tile + 1KB [16][32] P)
#define SMEM_BYTES 151552

__device__ __forceinline__ v8f zero8() {
  v8f z = {0.f,0.f,0.f,0.f,0.f,0.f,0.f,0.f};
  return z;
}

__device__ __forceinline__ v8f wmma_bf16(v16bf a, v16bf b, v8f c) {
  return __builtin_amdgcn_wmma_f32_16x16x32_bf16(false, a, false, b, (short)0, c,
                                                 false, false);
}

// sum across each 16-lane half; every lane in a half ends with that half's sum
__device__ __forceinline__ float hsum16(float x) {
  x += __shfl_xor(x, 1, 32);
  x += __shfl_xor(x, 2, 32);
  x += __shfl_xor(x, 4, 32);
  x += __shfl_xor(x, 8, 32);
  return x;
}

// --- fragment loaders (layouts per CDNA5 ISA 7.12.2) -------------------------
// A 16x32 bf16 from LDS row-major [16 rows][ldk]
__device__ __forceinline__ v16bf ldsA(const __bf16* base, int ldk, int kbase) {
  const int lane = threadIdx.x & 31;
  const __bf16* p = base + (lane & 15) * ldk + kbase + (lane >> 4) * 8;
  FragAB f;
  f.q[0] = *(const U128*)p;        // K 0..7   (or 8..15)
  f.q[1] = *(const U128*)(p + 16); // K 16..23 (or 24..31)
  return f.v;
}

// A 16x16 (upper K half zero), kbase multiple of 8
__device__ __forceinline__ v16bf ldsA16(const __bf16* base, int ldk, int kbase) {
  const int lane = threadIdx.x & 31;
  const __bf16* p = base + (lane & 15) * ldk + kbase + (lane >> 4) * 8;
  FragAB f;
  f.q[0] = *(const U128*)p;
  f.q[1].x[0] = 0; f.q[1].x[1] = 0; f.q[1].x[2] = 0; f.q[1].x[3] = 0;
  return f.v;
}

// B 32x16 bf16, operand stored as B^T row-major [N rows][ldk] (K contiguous)
__device__ __forceinline__ v16bf ldsB(const __bf16* base, int ldk, int n0, int kbase) {
  const int lane = threadIdx.x & 31;
  const __bf16* p = base + (n0 + (lane & 15)) * ldk + kbase + (lane >> 4) * 8;
  FragAB f;
  f.q[0] = *(const U128*)p;
  f.q[1] = *(const U128*)(p + 16);
  return f.v;
}

// B 16x16 (upper K half zero), from B^T row-major [N][ldk]
__device__ __forceinline__ v16bf ldsB16(const __bf16* base, int ldk, int n0, int kbase) {
  const int lane = threadIdx.x & 31;
  FragAB f;
  {
    const __bf16* p = base + (n0 + (lane & 15)) * ldk + kbase + (lane >> 4) * 8;
    f.q[0] = *(const U128*)p;
    #pragma unroll
    for (int i = 0; i < 4; i++) { f.q[1].x[i] = 0; }
  }
  return f.v;
}

// A 16x32 from GLOBAL f32 (v_cvt_pk_bf16_f32), rows >= rmax -> 0
__device__ __forceinline__ v16bf gldA(const float* __restrict__ X, int ldx, int row0,
                                      int rmax, int kbase) {
  const int lane = threadIdx.x & 31;
  const int row = row0 + (lane & 15);
  const int hf = lane >> 4;
  FragAB f;
  if (row < rmax) {
    const float* p = X + (size_t)row * ldx + kbase + hf * 8;
    #pragma unroll
    for (int i = 0; i < 8; i++) f.h[i] = (__bf16)p[i];
    #pragma unroll
    for (int i = 0; i < 8; i++) f.h[8 + i] = (__bf16)p[16 + i];
  } else {
    #pragma unroll
    for (int i = 0; i < 4; i++) { f.q[0].x[i] = 0; f.q[1].x[i] = 0; }
  }
  return f.v;
}

__device__ __forceinline__ void stage_w(const float* src, __bf16* dst, int n) {
  for (int i = threadIdx.x; i < n; i += 256) dst[i] = (__bf16)src[i];
}

// prefetch the 16 rows of tile t (global_prefetch_b8)
__device__ __forceinline__ void prefetch_tile(const float* __restrict__ X, int nrows, int t) {
  const int lane = threadIdx.x & 31;
  int row = t * 16 + (lane & 15);
  if (row < nrows) {
    const float* p = X + (size_t)row * D_MODEL + (lane >> 4) * 128;
    __builtin_prefetch(p, 0, 1);
    __builtin_prefetch(p + 64, 0, 1);
  }
}

// one 16-row tile of LN(X @ W^T + bias), bf16 to dst (ld=64); optional packed
// transposed copy to dstT [64][ldT]
__device__ __forceinline__ void proj_ln_tile(const float* __restrict__ X, int nrows, int t,
                                             const __bf16* Wlds, const float* bias,
                                             const float* gamma, const float* beta,
                                             __bf16* dst, int dstRow0,
                                             __bf16* dstT, int ldT, int rowT0) {
  const int lane = threadIdx.x & 31, ln = lane & 15, rb = (lane >> 4) * 8;
  v8f c[4];
  #pragma unroll
  for (int nt = 0; nt < 4; nt++) c[nt] = zero8();
  for (int s = 0; s < 8; s++) {  // K = 256, steps of 32
    v16bf a = gldA(X, D_MODEL, t * 16, nrows, s * 32);
    #pragma unroll
    for (int nt = 0; nt < 4; nt++)
      c[nt] = wmma_bf16(a, ldsB(Wlds, D_MODEL, nt * 16, s * 32), c[nt]);
  }
  #pragma unroll
  for (int nt = 0; nt < 4; nt++) {
    float bv = bias[nt * 16 + ln];
    #pragma unroll
    for (int r = 0; r < 8; r++) c[nt][r] += bv;
  }
  float mu[8], sc[8];
  #pragma unroll
  for (int r = 0; r < 8; r++) {
    float a1 = 0.f, a2 = 0.f;
    #pragma unroll
    for (int nt = 0; nt < 4; nt++) { float x = c[nt][r]; a1 += x; a2 += x * x; }
    float s1 = hsum16(a1) * (1.f / 64.f);
    float s2 = hsum16(a2) * (1.f / 64.f);
    mu[r] = s1;
    sc[r] = rsqrtf(s2 - s1 * s1 + LN_EPS);
  }
  #pragma unroll
  for (int nt = 0; nt < 4; nt++) {
    float g = gamma[nt * 16 + ln], bb = beta[nt * 16 + ln];
    Pack8 pk;
    #pragma unroll
    for (int r = 0; r < 8; r++) {
      float y = (c[nt][r] - mu[r]) * sc[r] * g + bb;
      dst[(dstRow0 + r + rb) * 64 + nt * 16 + ln] = (__bf16)y;
      pk.h[r] = (__bf16)y;
    }
    if (dstT)  // 8 consecutive rows per lane -> single ds_store_b128
      *(U128*)&dstT[(nt * 16 + ln) * ldT + rowT0 + rb] = pk.q;
  }
}

extern "C" __global__ __launch_bounds__(256)
void wda_fused(const float* __restrict__ turb, const float* __restrict__ wea,
               const float* w_t, const float* b_t, const float* w_w, const float* b_w,
               const float* g_t, const float* be_t, const float* g_w, const float* be_w,
               const float* w_in, const float* b_in, const float* w_ao, const float* b_ao,
               const float* w_op, const float* b_op, float* __restrict__ out) {
  __shared__ __align__(16) unsigned char smem[SMEM_BYTES];
  const int bt = blockIdx.x;             // one workgroup per (b,t)
  const int wave = threadIdx.x >> 5;
  const int lane = threadIdx.x & 31, ln = lane & 15, rb = (lane >> 4) * 8;

  __bf16* W0  = (__bf16*)(smem + OFF_W0);
  __bf16* HW  = (__bf16*)(smem + OFF_HW);
  __bf16* Q   = (__bf16*)(smem + OFF_Q);
  __bf16* K   = (__bf16*)(smem + OFF_K);   // k [208][64] (phase B onward)
  __bf16* HWT = (__bf16*)(smem + OFF_K);   // h_wT [64][112] (phases W/A only)
  __bf16* VT  = (__bf16*)(smem + OFF_V);   // vT [64][208]
  __bf16* stgT = (__bf16*)(smem + OFF_STG + wave * 3072); // [16][64]
  __bf16* stgP = stgT + 1024;                             // [16][32]

  const float* turbB = turb + (size_t)bt * NT * D_MODEL;
  const float* weaB  = wea  + (size_t)bt * NW * D_MODEL;

  // ---- Phase W: weather projection + LN (h_w row-major + transposed) ---
  stage_w(w_w, W0, D_HID * D_MODEL);
  __syncthreads();
  for (int t = wave; t < NTILES_W; t += 8) {
    prefetch_tile(weaB, NW, t);
    proj_ln_tile(weaB, NW, t, W0, b_w, g_w, be_w, HW, t * 16, HWT, 112, t * 16);
  }
  __syncthreads();
  stage_w(w_t, W0, D_HID * D_MODEL);
  __syncthreads();

  // ---- Phase A: h_t tile -> cross-attention (two-pass) -> ctx ----------
  for (int t = wave; t < NTILES_T; t += 8) {
    prefetch_tile(turbB, NT, t);
    if (t + 8 < NTILES_T) prefetch_tile(turbB, NT, t + 8);
    proj_ln_tile(turbB, NT, t, W0, b_t, g_t, be_t, stgT, 0, (__bf16*)0, 0, 0);
    v16bf a0 = ldsA(stgT, 64, 0), a1 = ldsA(stgT, 64, 32);

    // pass 1: row sums of exp(logits); only tile 6 needs the m-mask
    float rs[8];
    #pragma unroll
    for (int r = 0; r < 8; r++) rs[r] = 0.f;
    for (int mt = 0; mt < 6; mt++) {
      v8f s = zero8();
      s = wmma_bf16(a0, ldsB(HW, 64, mt * 16, 0), s);
      s = wmma_bf16(a1, ldsB(HW, 64, mt * 16, 32), s);
      #pragma unroll
      for (int r = 0; r < 8; r++) rs[r] += __expf(s[r] * 0.125f);
    }
    {
      v8f s = zero8();
      s = wmma_bf16(a0, ldsB(HW, 64, 96, 0), s);
      s = wmma_bf16(a1, ldsB(HW, 64, 96, 32), s);
      #pragma unroll
      for (int r = 0; r < 8; r++)
        rs[r] += (96 + ln < NW) ? __expf(s[r] * 0.125f) : 0.f;
    }
    #pragma unroll
    for (int r = 0; r < 8; r++) rs[r] = 1.f / hsum16(rs[r]);

    // pass 2: recompute P in 16x32 chunks; ctx += P @ h_w  (B from h_wT)
    v8f ctx[4];
    #pragma unroll
    for (int nt = 0; nt < 4; nt++) ctx[nt] = zero8();
    for (int mc = 0; mc < 3; mc++) {
      #pragma unroll
      for (int hh = 0; hh < 2; hh++) {
        int mt = mc * 2 + hh;
        v8f s = zero8();
        s = wmma_bf16(a0, ldsB(HW, 64, mt * 16, 0), s);
        s = wmma_bf16(a1, ldsB(HW, 64, mt * 16, 32), s);
        #pragma unroll
        for (int r = 0; r < 8; r++)
          stgP[(r + rb) * 32 + hh * 16 + ln] = (__bf16)__expf(s[r] * 0.125f);
      }
      v16bf pa = ldsA(stgP, 32, 0);
      #pragma unroll
      for (int nt = 0; nt < 4; nt++)
        ctx[nt] = wmma_bf16(pa, ldsB(HWT, 112, nt * 16, mc * 32), ctx[nt]);
    }
    {  // tail m-tile 6 (m = 96..111, masked), K=16 half-fragments
      v8f s = zero8();
      s = wmma_bf16(a0, ldsB(HW, 64, 96, 0), s);
      s = wmma_bf16(a1, ldsB(HW, 64, 96, 32), s);
      #pragma unroll
      for (int r = 0; r < 8; r++)
        stgP[(r + rb) * 32 + ln] =
            (__bf16)((96 + ln < NW) ? __expf(s[r] * 0.125f) : 0.f);
      v16bf pa = ldsA16(stgP, 32, 0);
      #pragma unroll
      for (int nt = 0; nt < 4; nt++)
        ctx[nt] = wmma_bf16(pa, ldsB16(HWT, 112, nt * 16, 96), ctx[nt]);
    }
    #pragma unroll
    for (int nt = 0; nt < 4; nt++)
      #pragma unroll
      for (int r = 0; r < 8; r++)
        Q[(t * 16 + r + rb) * 64 + nt * 16 + ln] = (__bf16)(ctx[nt][r] * rs[r]);
  }
  __syncthreads();

  // ---- Phase B: q,k,v = ctx @ in_proj^T (v stored transposed) ----------
  stage_w(w_in, W0, 192 * 64);
  stage_w(w_ao, W0 + 192 * 64, 64 * 64);
  __syncthreads();
  for (int t = wave; t < NTILES_T; t += 8) {
    v16bf a0 = ldsA(Q + t * 16 * 64, 64, 0), a1 = ldsA(Q + t * 16 * 64, 64, 32);
    #pragma unroll
    for (int w3 = 0; w3 < 2; w3++) {   // q overwrites ctx in place; then k
      __bf16* dst = (w3 == 0) ? Q : K;
      #pragma unroll
      for (int nt = 0; nt < 4; nt++) {
        v8f c = zero8();
        c = wmma_bf16(a0, ldsB(W0, 64, w3 * 64 + nt * 16, 0), c);
        c = wmma_bf16(a1, ldsB(W0, 64, w3 * 64 + nt * 16, 32), c);
        float bv = b_in[w3 * 64 + nt * 16 + ln];
        #pragma unroll
        for (int r = 0; r < 8; r++)
          dst[(t * 16 + r + rb) * 64 + nt * 16 + ln] = (__bf16)(c[r] + bv);
      }
    }
    #pragma unroll
    for (int nt = 0; nt < 4; nt++) {   // v -> vT [64][208], packed b128 stores
      v8f c = zero8();
      c = wmma_bf16(a0, ldsB(W0, 64, 128 + nt * 16, 0), c);
      c = wmma_bf16(a1, ldsB(W0, 64, 128 + nt * 16, 32), c);
      float bv = b_in[128 + nt * 16 + ln];
      Pack8 pk;
      #pragma unroll
      for (int r = 0; r < 8; r++) pk.h[r] = (__bf16)(c[r] + bv);
      *(U128*)&VT[(nt * 16 + ln) * 208 + t * 16 + rb] = pk.q;
    }
  }
  __syncthreads();

  // ---- Phase C: self-attention (4 heads, hd=16, two-pass) + attn_out ----
  const __bf16* Wao = W0 + 192 * 64;
  for (int t = wave; t < NTILES_T; t += 8) {
    #pragma unroll
    for (int h = 0; h < 4; h++) {
      v16bf qa = ldsA16(Q + t * 16 * 64, 64, h * 16);
      float rs[8];
      #pragma unroll
      for (int r = 0; r < 8; r++) rs[r] = 0.f;
      for (int mt = 0; mt < 12; mt++) {
        v8f s = zero8();
        s = wmma_bf16(qa, ldsB16(K, 64, mt * 16, h * 16), s);
        #pragma unroll
        for (int r = 0; r < 8; r++) rs[r] += __expf(s[r] * 0.25f);
      }
      {
        v8f s = zero8();
        s = wmma_bf16(qa, ldsB16(K, 64, 192, h * 16), s);
        #pragma unroll
        for (int r = 0; r < 8; r++)
          rs[r] += (192 + ln < NT) ? __expf(s[r] * 0.25f) : 0.f;
      }
      #pragma unroll
      for (int r = 0; r < 8; r++) rs[r] = 1.f / hsum16(rs[r]);

      v8f o = zero8();
      for (int mc = 0; mc < 6; mc++) {
        #pragma unroll
        for (int hh = 0; hh < 2; hh++) {
          int mt = mc * 2 + hh;
          v8f s = zero8();
          s = wmma_bf16(qa, ldsB16(K, 64, mt * 16, h * 16), s);
          #pragma unroll
          for (int r = 0; r < 8; r++)
            stgP[(r + rb) * 32 + hh * 16 + ln] = (__bf16)__expf(s[r] * 0.25f);
        }
        v16bf pa = ldsA(stgP, 32, 0);
        o = wmma_bf16(pa, ldsB(VT, 208, h * 16, mc * 32), o);
      }
      {  // tail m-tile 12 (m = 192..207, masked), K=16 half-fragments
        v8f s = zero8();
        s = wmma_bf16(qa, ldsB16(K, 64, 192, h * 16), s);
        #pragma unroll
        for (int r = 0; r < 8; r++)
          stgP[(r + rb) * 32 + ln] =
              (__bf16)((192 + ln < NT) ? __expf(s[r] * 0.25f) : 0.f);
        v16bf pa = ldsA16(stgP, 32, 0);
        o = wmma_bf16(pa, ldsB16(VT, 208, h * 16, 192), o);
      }
      #pragma unroll
      for (int r = 0; r < 8; r++)
        stgT[(r + rb) * 64 + h * 16 + ln] = (__bf16)(o[r] * rs[r]);
    }
    // attn_out: enh = o @ attn_out_w^T + b  (overwrites own q rows)
    v16bf oa0 = ldsA(stgT, 64, 0), oa1 = ldsA(stgT, 64, 32);
    #pragma unroll
    for (int nt = 0; nt < 4; nt++) {
      v8f c = zero8();
      c = wmma_bf16(oa0, ldsB(Wao, 64, nt * 16, 0), c);
      c = wmma_bf16(oa1, ldsB(Wao, 64, nt * 16, 32), c);
      float bv = b_ao[nt * 16 + ln];
      #pragma unroll
      for (int r = 0; r < 8; r++)
        Q[(t * 16 + r + rb) * 64 + nt * 16 + ln] = (__bf16)(c[r] + bv);
    }
  }
  __syncthreads();

  // ---- Phase F: out = turbine + enh @ out_proj^T + b -------------------
  stage_w(w_op, W0, 256 * 64);
  __syncthreads();
  for (int t = wave; t < NTILES_T; t += 8) {
    v16bf a0 = ldsA(Q + t * 16 * 64, 64, 0), a1 = ldsA(Q + t * 16 * 64, 64, 32);
    for (int nt = 0; nt < 16; nt++) {
      v8f c = zero8();
      c = wmma_bf16(a0, ldsB(W0, 64, nt * 16, 0), c);
      c = wmma_bf16(a1, ldsB(W0, 64, nt * 16, 32), c);
      int col = nt * 16 + ln;
      float bo = b_op[col];
      #pragma unroll
      for (int r = 0; r < 8; r++) {
        int row = t * 16 + r + rb;
        if (row < NT) {
          size_t idx = (size_t)bt * NT * D_MODEL + (size_t)row * D_MODEL + col;
          // output is write-once: non-temporal store keeps L2 for input reuse
          __builtin_nontemporal_store(turb[idx] + c[r] + bo, &out[idx]);
        }
      }
    }
  }
}

extern "C" void kernel_launch(void* const* d_in, const int* in_sizes, int n_in,
                              void* d_out, int out_size, void* d_ws, size_t ws_size,
                              hipStream_t stream) {
  (void)in_sizes; (void)n_in; (void)out_size; (void)d_ws; (void)ws_size;
  wda_fused<<<dim3(768), dim3(256), 0, stream>>>(
      (const float*)d_in[0],  (const float*)d_in[1],  (const float*)d_in[2],
      (const float*)d_in[3],  (const float*)d_in[4],  (const float*)d_in[5],
      (const float*)d_in[6],  (const float*)d_in[7],  (const float*)d_in[8],
      (const float*)d_in[9],  (const float*)d_in[10], (const float*)d_in[11],
      (const float*)d_in[12], (const float*)d_in[13], (const float*)d_in[14],
      (const float*)d_in[15], (float*)d_out);
}
